// LSTM_33990371181093
// MI455X (gfx1250) — matrix operand order repaired and checked by reading, not compiled, not run on hardware
//
#include <hip/hip_runtime.h>
#include <hip/hip_bf16.h>

// Problem constants (match reference)
constexpr int Bb = 64;    // batch
constexpr int Tt = 512;   // time steps
constexpr int Ii = 256;   // input dim
constexpr int Hh = 512;   // hidden dim
constexpr int Cc = 128;   // output classes
constexpr int NWG = 16;   // persistent workgroups for recurrence (<= #WGPs)
constexpr int SWP = Hh + 8;  // padded LDS row stride (f16) -> conflict-free b128

typedef __attribute__((ext_vector_type(16))) _Float16 v16h;
typedef __attribute__((ext_vector_type(8)))  _Float16 v8h;
typedef __attribute__((ext_vector_type(8)))  float    v8f;
typedef __attribute__((ext_vector_type(4)))  int      v4i;

// ---------------------------------------------------------------------------
// Async global -> LDS copy (gfx1250 GLOBAL_LOAD_ASYNC_TO_LDS_B128, ASYNCcnt).
// Builtin signature (from clang diagnostic): (v4i addrspace(1)*,
// v4i addrspace(3)*, imm offset, imm cpol). Guarded by __has_builtin;
// fallback is a plain load + ds_store round trip.
// ---------------------------------------------------------------------------
#if defined(__AMDGCN__) && __has_builtin(__builtin_amdgcn_global_load_async_to_lds_b128)
#define HAVE_ASYNC_LDS 1
#else
#define HAVE_ASYNC_LDS 0
#endif

#if HAVE_ASYNC_LDS
typedef __attribute__((address_space(1))) v4i GlobV4i;
typedef __attribute__((address_space(3))) v4i LdsV4i;
#endif

__device__ __forceinline__ void async_copy16(const _Float16* g, _Float16* l) {
#if HAVE_ASYNC_LDS
  __builtin_amdgcn_global_load_async_to_lds_b128((GlobV4i*)g, (LdsV4i*)l, 0, 0);
#else
  *(v8h*)l = *(const v8h*)g;
#endif
}

__device__ __forceinline__ void async_wait_all() {
#if HAVE_ASYNC_LDS
#if __has_builtin(__builtin_amdgcn_s_wait_asynccnt)
  __builtin_amdgcn_s_wait_asynccnt(0);
#else
  asm volatile("s_wait_asynccnt 0x0" ::: "memory");
#endif
#endif
}

// ---------------------------------------------------------------------------
// WMMA fragment loaders (v_wmma_f32_16x16x32_f16 layouts, CDNA5 ISA)
// A (16x32, MxK): lane = half*16 + m ; k = half*8 + e (e<8), 16+half*8+(e-8)
// B (32x16, KxN) with W stored [N][K]: lane = half*16 + n ; k = half*8 + e (e<8), 16+half*8+(e-8)
// Both are contiguous 16B accesses per lane.
// ---------------------------------------------------------------------------
__device__ __forceinline__ v16h load_afrag(const _Float16* rowptr, int k, int half) {
  const _Float16* p = rowptr + k + half * 8;
  v8h lo = *(const v8h*)p;
  v8h hi = *(const v8h*)(p + 16);
  v16h r;
#pragma unroll
  for (int i = 0; i < 8; ++i) { r[i] = lo[i]; r[i + 8] = hi[i]; }
  return r;
}

__device__ __forceinline__ v16h load_bfrag(const _Float16* rowptr, int k, int half) {
  const _Float16* p = rowptr + k + half * 8;
  v8h lo = *(const v8h*)p;
  v8h hi = *(const v8h*)(p + 16);
  v16h r;
#pragma unroll
  for (int i = 0; i < 8; ++i) { r[i] = lo[i]; r[i + 8] = hi[i]; }
  return r;
}

// ---------------------------------------------------------------------------
// f32 -> f16 conversion (grid-stride)
// ---------------------------------------------------------------------------
__global__ void cvt_f16_kernel(_Float16* __restrict__ dst,
                               const float* __restrict__ src, size_t n) {
  size_t i = (size_t)blockIdx.x * blockDim.x + threadIdx.x;
  size_t stride = (size_t)gridDim.x * blockDim.x;
  for (; i < n; i += stride) dst[i] = (_Float16)src[i];
}

// ---------------------------------------------------------------------------
// Out[M,N] = A[M,K] * W[N,K]^T + bias0[N] (+ bias1[N])
// 8 waves arranged 4(M) x 2(N); wave tile 32M x 64N (8 accumulators).
// Register double-buffered k pipeline (k+32 fragments issued before the
// 8 WMMAs of k) -> partial s_wait_loadcnt, loads overlap the XDL pipe.
// ---------------------------------------------------------------------------
__global__ __launch_bounds__(256) void wmma_gemm_bias(
    float* __restrict__ Out, const _Float16* __restrict__ A,
    const _Float16* __restrict__ W, const float* __restrict__ bias0,
    const float* __restrict__ bias1, int M, int N, int K) {
  const int lane = threadIdx.x & 31;
  const int wave = threadIdx.x >> 5;
  const int half = lane >> 4;
  const int l16  = lane & 15;
  const int mw   = wave & 3;
  const int nw   = wave >> 2;
  const int mbase = blockIdx.y * 128 + mw * 32;
  const int nbase = blockIdx.x * 128 + nw * 64;
  (void)M;

  v8f acc[2][4];
#pragma unroll
  for (int j = 0; j < 4; ++j) {
    int n = nbase + j * 16 + l16;
    float bv = bias0 ? bias0[n] : 0.0f;
    if (bias1) bv += bias1[n];
#pragma unroll
    for (int i = 0; i < 2; ++i)
#pragma unroll
      for (int r = 0; r < 8; ++r) acc[i][j][r] = bv;
  }

  const _Float16* arow0 = A + (size_t)(mbase + l16) * K;
  const _Float16* arow1 = A + (size_t)(mbase + 16 + l16) * K;
  const _Float16* brow[4];
#pragma unroll
  for (int j = 0; j < 4; ++j) brow[j] = W + (size_t)(nbase + j * 16 + l16) * K;

  const int iters = K >> 5;
  v16h af[2][2], bf[2][4];
  af[0][0] = load_afrag(arow0, 0, half);
  af[0][1] = load_afrag(arow1, 0, half);
#pragma unroll
  for (int j = 0; j < 4; ++j) bf[0][j] = load_bfrag(brow[j], 0, half);

#pragma unroll 2
  for (int kk = 0; kk < iters - 1; ++kk) {
    const int cur = kk & 1, nxt = cur ^ 1;
    const int k = (kk + 1) << 5;
    af[nxt][0] = load_afrag(arow0, k, half);
    af[nxt][1] = load_afrag(arow1, k, half);
#pragma unroll
    for (int j = 0; j < 4; ++j) bf[nxt][j] = load_bfrag(brow[j], k, half);
#pragma unroll
    for (int i = 0; i < 2; ++i)
#pragma unroll
      for (int j = 0; j < 4; ++j)
        acc[i][j] = __builtin_amdgcn_wmma_f32_16x16x32_f16(
            false, af[cur][i], false, bf[cur][j], (short)0, acc[i][j], false, false);
  }
  {
    const int cur = (iters - 1) & 1;
#pragma unroll
    for (int i = 0; i < 2; ++i)
#pragma unroll
      for (int j = 0; j < 4; ++j)
        acc[i][j] = __builtin_amdgcn_wmma_f32_16x16x32_f16(
            false, af[cur][i], false, bf[cur][j], (short)0, acc[i][j], false, false);
  }

#pragma unroll
  for (int i = 0; i < 2; ++i)
#pragma unroll
    for (int j = 0; j < 4; ++j) {
      int n = nbase + j * 16 + l16;
#pragma unroll
      for (int r = 0; r < 8; ++r) {
        int m = mbase + i * 16 + half * 8 + r;
        Out[(size_t)m * N + n] = acc[i][j][r];
      }
    }
}

// ---------------------------------------------------------------------------
// Per-layer state init + barrier counter reset.
// ---------------------------------------------------------------------------
__global__ void init_state_kernel(_Float16* __restrict__ h16,
                                  float* __restrict__ c,
                                  const float* __restrict__ h0,
                                  const float* __restrict__ c0, int layer,
                                  unsigned* __restrict__ counter) {
  int i = blockIdx.x * blockDim.x + threadIdx.x;
  if (i == 0) *counter = 0u;
  if (i < Bb * Hh) {
    h16[i] = (_Float16)h0[(size_t)layer * Bb * Hh + i];
    c[i]   = c0[(size_t)layer * Bb * Hh + i];
  }
}

// ---------------------------------------------------------------------------
// Persistent LSTM recurrence. 16 WGs; WG wg owns hidden units [wg*32, +32).
// Per step: [64 x 128] GEMM, K=512. W_hh slice LDS-resident (async-staged
// once); h_prev async-staged into LDS each step (issued right after the grid
// barrier, overlapped with xW prefetch + acc init). Both WMMA operands then
// come from bank-conflict-free padded LDS. Grid barrier between steps.
// ---------------------------------------------------------------------------
__global__ __launch_bounds__(256) void lstm_recurrence_kernel(
    const float* __restrict__ xW,      // [B*T, 4H] gate pre-acts (biases folded)
    const _Float16* __restrict__ Whh,  // [4H, H] f16
    _Float16* __restrict__ hbuf,       // [2, B*H] f16 ping-pong
    float* __restrict__ cbuf,          // [B*H] f32 cell state
    _Float16* __restrict__ hs_out,     // [B*T, H] f16 hidden sequence
    unsigned* __restrict__ counter) {
  extern __shared__ char smem[];
  _Float16* sW = (_Float16*)smem;                                     // [128][SWP]
  float* sG = (float*)(smem + (size_t)128 * SWP * sizeof(_Float16));  // [64][132]
  _Float16* sH = (_Float16*)(smem + (size_t)128 * SWP * sizeof(_Float16) +
                             (size_t)64 * 132 * sizeof(float));       // [64][SWP]
  const int GP = 132;

  const int j0   = blockIdx.x * 32;
  const int lane = threadIdx.x & 31;
  const int wave = threadIdx.x >> 5;
  const int half = lane >> 4;
  const int l16  = lane & 15;
  const int mtile = wave & 3;   // 4 x 16 = 64 batch rows
  const int ngrp  = wave >> 2;  // 2 x 64 = 128 gate cols

  // One-shot async stage of this WG's W_hh slice into LDS (padded rows).
  for (int idx = threadIdx.x; idx < 128 * (Hh / 8); idx += 256) {
    int row = idx >> 6;
    int off = (idx & 63) * 8;
    int gate = row >> 5, jl = row & 31;
    async_copy16(Whh + ((size_t)(gate * Hh + j0 + jl)) * Hh + off,
                 sW + (size_t)row * SWP + off);
  }
  async_wait_all();
  __syncthreads();

  const _Float16* sbrow[4];
#pragma unroll
  for (int j = 0; j < 4; ++j)
    sbrow[j] = sW + (size_t)(ngrp * 64 + j * 16 + l16) * SWP;
  const _Float16* sarow = sH + (size_t)(mtile * 16 + l16) * SWP;

  for (int t = 0; t < Tt; ++t) {
    const _Float16* hprev = hbuf + (size_t)(t & 1) * Bb * Hh;
    _Float16* hnext = hbuf + (size_t)((t & 1) ^ 1) * Bb * Hh;

    // Kick off async h_prev -> LDS stage (64 KB) immediately after barrier.
    for (int idx = threadIdx.x; idx < 64 * (Hh / 8); idx += 256) {
      int row = idx >> 6;
      int off = (idx & 63) * 8;
      async_copy16(hprev + (size_t)row * Hh + off, sH + (size_t)row * SWP + off);
    }

    // Overlap: prefetch this step's xW gate rows for the pointwise stage.
#pragma unroll
    for (int i = 0; i < 8; ++i) {
      int idx = threadIdx.x + i * 256;
      int b = idx >> 5, jl = idx & 31;
      const float* xw = xW + ((size_t)b * Tt + t) * (4 * Hh) + j0 + jl;
      __builtin_prefetch(xw + 0 * Hh, 0, 1);
      __builtin_prefetch(xw + 1 * Hh, 0, 1);
      __builtin_prefetch(xw + 2 * Hh, 0, 1);
      __builtin_prefetch(xw + 3 * Hh, 0, 1);
    }

    v8f acc[4];
#pragma unroll
    for (int j = 0; j < 4; ++j)
#pragma unroll
      for (int r = 0; r < 8; ++r) acc[j][r] = 0.0f;

    async_wait_all();
    __syncthreads();

    // K pipeline: both operands from LDS, register double-buffered.
    constexpr int iters = Hh >> 5;  // 16
    v16h af[2], bf[2][4];
    af[0] = load_afrag(sarow, 0, half);
#pragma unroll
    for (int j = 0; j < 4; ++j) bf[0][j] = load_bfrag(sbrow[j], 0, half);

#pragma unroll 2
    for (int kk = 0; kk < iters - 1; ++kk) {
      const int cur = kk & 1, nxt = cur ^ 1;
      const int k = (kk + 1) << 5;
      af[nxt] = load_afrag(sarow, k, half);
#pragma unroll
      for (int j = 0; j < 4; ++j) bf[nxt][j] = load_bfrag(sbrow[j], k, half);
#pragma unroll
      for (int j = 0; j < 4; ++j)
        acc[j] = __builtin_amdgcn_wmma_f32_16x16x32_f16(
            false, af[cur], false, bf[cur][j], (short)0, acc[j], false, false);
    }
    {
      const int cur = (iters - 1) & 1;
#pragma unroll
      for (int j = 0; j < 4; ++j)
        acc[j] = __builtin_amdgcn_wmma_f32_16x16x32_f16(
            false, af[cur], false, bf[cur][j], (short)0, acc[j], false, false);
    }

    // Stage gates to LDS (unit j's 4 gates live in different tiles/waves).
#pragma unroll
    for (int j = 0; j < 4; ++j) {
      int n = ngrp * 64 + j * 16 + l16;
#pragma unroll
      for (int r = 0; r < 8; ++r) {
        int b = mtile * 16 + half * 8 + r;
        sG[b * GP + n] = acc[j][r];
      }
    }
    __syncthreads();

    // Pointwise LSTM cell: 64 x 32 = 2048 elems, 8 per thread.
#pragma unroll
    for (int i = 0; i < 8; ++i) {
      int idx = threadIdx.x + i * 256;
      int b = idx >> 5, jl = idx & 31;
      size_t mrow = (size_t)b * Tt + t;
      const float* xw = xW + mrow * (4 * Hh) + j0 + jl;
      float gi = sG[b * GP + 0 * 32 + jl] + xw[0 * Hh];
      float gf = sG[b * GP + 1 * 32 + jl] + xw[1 * Hh];
      float gg = sG[b * GP + 2 * 32 + jl] + xw[2 * Hh];
      float go = sG[b * GP + 3 * 32 + jl] + xw[3 * Hh];
      float si = 1.0f / (1.0f + __expf(-gi));
      float sf = 1.0f / (1.0f + __expf(-gf));
      float tg = tanhf(gg);
      float so = 1.0f / (1.0f + __expf(-go));
      size_t cidx = (size_t)b * Hh + j0 + jl;
      float c = sf * cbuf[cidx] + si * tg;
      cbuf[cidx] = c;
      float h = so * tanhf(c);
      _Float16 h16 = (_Float16)h;
      hnext[cidx] = h16;
      hs_out[mrow * Hh + j0 + jl] = h16;
    }

    // Grid-wide barrier before the next step consumes hnext.
    __threadfence();
    __syncthreads();
    if (threadIdx.x == 0) {
      atomicAdd(counter, 1u);
      unsigned target = (unsigned)(t + 1) * NWG;
      while (__hip_atomic_load(counter, __ATOMIC_ACQUIRE,
                               __HIP_MEMORY_SCOPE_AGENT) < target) {
        __builtin_amdgcn_s_sleep(1);
      }
    }
    __syncthreads();
  }
}

// ---------------------------------------------------------------------------
// FC head on last timestep: out[b, c] = hs1[b*T + T-1, :] @ Wfc[c, :] + bfc[c]
// ---------------------------------------------------------------------------
__global__ __launch_bounds__(256) void fc_head_kernel(
    const _Float16* __restrict__ hs1, const _Float16* __restrict__ Wfc,
    const float* __restrict__ bfc, float* __restrict__ out) {
  const int lane = threadIdx.x & 31;
  const int wave = threadIdx.x >> 5;
  const int half = lane >> 4;
  const int l16  = lane & 15;
  const int mtile = wave & 3;
  const int ngrp  = wave >> 2;

  v8f acc[4];
#pragma unroll
  for (int j = 0; j < 4; ++j) {
    float bv = bfc[ngrp * 64 + j * 16 + l16];
#pragma unroll
    for (int r = 0; r < 8; ++r) acc[j][r] = bv;
  }

  int brow = mtile * 16 + l16;
  const _Float16* arow = hs1 + ((size_t)brow * Tt + (Tt - 1)) * Hh;
  for (int k = 0; k < Hh; k += 32) {
    v16h a = load_afrag(arow, k, half);
#pragma unroll
    for (int j = 0; j < 4; ++j) {
      v16h b = load_bfrag(Wfc + (size_t)(ngrp * 64 + j * 16 + l16) * Hh, k, half);
      acc[j] = __builtin_amdgcn_wmma_f32_16x16x32_f16(
          false, a, false, b, (short)0, acc[j], false, false);
    }
  }

#pragma unroll
  for (int j = 0; j < 4; ++j) {
    int n = ngrp * 64 + j * 16 + l16;
#pragma unroll
    for (int r = 0; r < 8; ++r) {
      int b = mtile * 16 + half * 8 + r;
      out[(size_t)b * Cc + n] = acc[j][r];
    }
  }
}

// ---------------------------------------------------------------------------
extern "C" void kernel_launch(void* const* d_in, const int* in_sizes, int n_in,
                              void* d_out, int out_size, void* d_ws,
                              size_t ws_size, hipStream_t stream) {
  (void)in_sizes; (void)n_in; (void)out_size; (void)ws_size;

  const float* x    = (const float*)d_in[0];
  const float* h0   = (const float*)d_in[1];
  const float* c0   = (const float*)d_in[2];
  const float* Wih0 = (const float*)d_in[3];
  const float* Whh0 = (const float*)d_in[4];
  const float* bih0 = (const float*)d_in[5];
  const float* bhh0 = (const float*)d_in[6];
  const float* Wih1 = (const float*)d_in[7];
  const float* Whh1 = (const float*)d_in[8];
  const float* bih1 = (const float*)d_in[9];
  const float* bhh1 = (const float*)d_in[10];
  const float* Wfc  = (const float*)d_in[11];
  const float* bfc  = (const float*)d_in[12];

  char* w = (char*)d_ws;
  size_t off = 0;
  auto carve = [&](size_t bytes) {
    void* p = w + off;
    off = (off + bytes + 255) & ~(size_t)255;
    return p;
  };
  float*     xW     = (float*)    carve((size_t)Bb * Tt * 4 * Hh * 4);  // reused per layer
  _Float16*  x16    = (_Float16*) carve((size_t)Bb * Tt * Ii * 2);
  _Float16*  wih0_h = (_Float16*) carve((size_t)4 * Hh * Ii * 2);
  _Float16*  whh0_h = (_Float16*) carve((size_t)4 * Hh * Hh * 2);
  _Float16*  wih1_h = (_Float16*) carve((size_t)4 * Hh * Hh * 2);
  _Float16*  whh1_h = (_Float16*) carve((size_t)4 * Hh * Hh * 2);
  _Float16*  wfc_h  = (_Float16*) carve((size_t)Cc * Hh * 2);
  _Float16*  hs0    = (_Float16*) carve((size_t)Bb * Tt * Hh * 2);
  _Float16*  hs1    = (_Float16*) carve((size_t)Bb * Tt * Hh * 2);
  _Float16*  hbuf   = (_Float16*) carve((size_t)2 * Bb * Hh * 2);
  float*     cbuf   = (float*)    carve((size_t)Bb * Hh * 4);
  unsigned*  counter= (unsigned*) carve(256);

  const int recur_smem = 128 * SWP * (int)sizeof(_Float16) +
                         64 * 132 * (int)sizeof(float) +
                         64 * SWP * (int)sizeof(_Float16);
  static bool attr_set = false;
  if (!attr_set) {  // host-side attribute, not a stream op
    (void)hipFuncSetAttribute((const void*)lstm_recurrence_kernel,
                              hipFuncAttributeMaxDynamicSharedMemorySize,
                              recur_smem);
    attr_set = true;
  }

  // 1) f32 -> f16 conversions
  cvt_f16_kernel<<<512, 256, 0, stream>>>(x16, x, (size_t)Bb * Tt * Ii);
  cvt_f16_kernel<<<256, 256, 0, stream>>>(wih0_h, Wih0, (size_t)4 * Hh * Ii);
  cvt_f16_kernel<<<256, 256, 0, stream>>>(whh0_h, Whh0, (size_t)4 * Hh * Hh);
  cvt_f16_kernel<<<256, 256, 0, stream>>>(wih1_h, Wih1, (size_t)4 * Hh * Hh);
  cvt_f16_kernel<<<256, 256, 0, stream>>>(whh1_h, Whh1, (size_t)4 * Hh * Hh);
  cvt_f16_kernel<<<64, 256, 0, stream>>>(wfc_h, Wfc, (size_t)Cc * Hh);

  // 2) Layer 0
  wmma_gemm_bias<<<dim3(4 * Hh / 128, Bb * Tt / 128), 256, 0, stream>>>(
      xW, x16, wih0_h, bih0, bhh0, Bb * Tt, 4 * Hh, Ii);
  init_state_kernel<<<(Bb * Hh + 255) / 256, 256, 0, stream>>>(
      hbuf, cbuf, h0, c0, 0, counter);
  lstm_recurrence_kernel<<<NWG, 256, recur_smem, stream>>>(
      xW, whh0_h, hbuf, cbuf, hs0, counter);

  // 3) Layer 1
  wmma_gemm_bias<<<dim3(4 * Hh / 128, Bb * Tt / 128), 256, 0, stream>>>(
      xW, hs0, wih1_h, bih1, bhh1, Bb * Tt, 4 * Hh, Hh);
  init_state_kernel<<<(Bb * Hh + 255) / 256, 256, 0, stream>>>(
      hbuf, cbuf, h0, c0, 1, counter);
  lstm_recurrence_kernel<<<NWG, 256, recur_smem, stream>>>(
      xW, whh1_h, hbuf, cbuf, hs1, counter);

  // 4) FC head
  fc_head_kernel<<<1, 256, 0, stream>>>(hs1, wfc_h, bfc, (float*)d_out);
}
